// HeteroGraphClassifier_12790412608057
// MI455X (gfx1250) — compile-verified, matrix operand order checked by reading, not run on hardware
//
#include <hip/hip_runtime.h>

// ---------------------------------------------------------------------------
// HeteroGraphClassifier forward on gfx1250 (MI455X).
// Dense GEMMs -> v_wmma_f32_16x16x32_f16 (f32->f16 LDS staging, f32 accum),
// software-pipelined global->reg->LDS staging, vectorized b128 loads.
// Segment means -> global_atomic_add_f32 scatter + divide.
// Concats folded into accumulate-GEMMs; dead h2v branch eliminated.
// NOTE: all GEMM call sites have M % 64 == 0 (no M guard needed).
// ---------------------------------------------------------------------------

typedef __attribute__((ext_vector_type(16))) _Float16 v16h;
typedef __attribute__((ext_vector_type(8)))  _Float16 v8h;
typedef __attribute__((ext_vector_type(8)))  float    v8f;

#define BM 64
#define BN 64
#define BK 32

#define F_ACC   1   // C += tile (read existing C)
#define F_BN    2   // batchnorm epilogue
#define F_LRELU 4   // leaky relu 0.01
#define F_RES   8   // add residual (after BN, before lrelu)

// Stage one 8-element strip of A and one of W into registers (no exec-mask
// predication: tails use clamped addresses + select).
__device__ __forceinline__ void stage_load(
    const float* __restrict__ A, const float* __restrict__ W,
    int N, int K, int m0, int n0, int k0,
    int arow, int acol, int bk, int bnc,
    float (&ar)[8], float (&br)[8])
{
    const bool fullK = (k0 + BK <= K);
    // ---- A strip: row m0+arow, cols k0+acol .. +7 ----
    {
        const float* ap = A + (size_t)(m0 + arow) * K + (size_t)(k0 + acol);
        if (fullK && ((K & 3) == 0)) {
            const float4 x = *(const float4*)ap;
            const float4 y = *(const float4*)(ap + 4);
            ar[0] = x.x; ar[1] = x.y; ar[2] = x.z; ar[3] = x.w;
            ar[4] = y.x; ar[5] = y.y; ar[6] = y.z; ar[7] = y.w;
        } else if (fullK) {
            #pragma unroll
            for (int j = 0; j < 8; ++j) ar[j] = ap[j];
        } else {
            const size_t base = (size_t)(m0 + arow) * K;
            #pragma unroll
            for (int j = 0; j < 8; ++j) {
                const int gk = k0 + acol + j;
                const int ck = gk < K ? gk : K - 1;   // clamp, no branch
                const float v = A[base + ck];
                ar[j] = gk < K ? v : 0.0f;
            }
        }
    }
    // ---- B strip: row k0+bk of W, cols n0+bnc .. +7 ----
    {
        const int gk = k0 + bk;
        if (fullK && (n0 + BN <= N) && ((N & 3) == 0)) {
            const float* bp = W + (size_t)gk * N + (size_t)(n0 + bnc);
            const float4 x = *(const float4*)bp;
            const float4 y = *(const float4*)(bp + 4);
            br[0] = x.x; br[1] = x.y; br[2] = x.z; br[3] = x.w;
            br[4] = y.x; br[5] = y.y; br[6] = y.z; br[7] = y.w;
        } else {
            const int ckr = gk < K ? gk : K - 1;
            const size_t base = (size_t)ckr * N;
            #pragma unroll
            for (int j = 0; j < 8; ++j) {
                const int gn = n0 + bnc + j;
                const int cn = gn < N ? gn : N - 1;   // clamp, no branch
                const float v = W[base + cn];
                br[j] = (gk < K && gn < N) ? v : 0.0f;
            }
        }
    }
}

__global__ __launch_bounds__(256)
void gemm_wmma(const float* __restrict__ A, const float* __restrict__ W,
               const float* __restrict__ bias,
               const float* __restrict__ bng, const float* __restrict__ bnb,
               const float* __restrict__ bnm, const float* __restrict__ bnv,
               const float* __restrict__ Res, float* __restrict__ C,
               int M, int N, int K, int flags)
{
    // C[M,N] (+)= A[M,K] @ W[K,N]  with fused epilogue.  Requires M % 64 == 0.
    __shared__ __align__(32) _Float16 As[BM][BK];   // row-major tile
    __shared__ __align__(32) _Float16 Bs[BN][BK];   // W tile, transposed (n,k)

    const int tid = threadIdx.x;
    const int m0 = blockIdx.y * BM;
    const int n0 = blockIdx.x * BN;

    const int wave = tid >> 5;     // 8 waves
    const int lane = tid & 31;
    const int wm = wave & 3;       // 4 row-tiles of 16
    const int wn = wave >> 2;      // 2 col-groups of 32
    const int lr = lane & 15;
    const int lh = lane >> 4;      // K-half selector (0 or 1)

    // staging indices: A -> 256 thr * 8 elems = 64x32 ; B -> 32(k) x 64(n)
    const int arow = tid >> 2;
    const int acol = (tid & 3) * 8;
    const int bk   = tid & 31;
    const int bnc  = (tid >> 5) * 8;

    v8f acc0 = {};
    v8f acc1 = {};

    float ar[8], br[8];
    stage_load(A, W, N, K, m0, n0, 0, arow, acol, bk, bnc, ar, br);

    const int ktiles = (K + BK - 1) / BK;
    for (int kt = 0; kt < ktiles; ++kt) {
        // commit staged registers to LDS
        {
            v8h ah;
            #pragma unroll
            for (int j = 0; j < 8; ++j) ah[j] = (_Float16)ar[j];
            *(v8h*)&As[arow][acol] = ah;            // one 16B ds_store
            #pragma unroll
            for (int j = 0; j < 8; ++j)
                Bs[bnc + j][bk] = (_Float16)br[j];  // transposed scatter
        }
        __syncthreads();

        // prefetch / stage next tile while WMMAs run
        if (kt + 1 < ktiles)
            stage_load(A, W, N, K, m0, n0, (kt + 1) * BK,
                       arow, acol, bk, bnc, ar, br);
        if (kt + 2 < ktiles) {
            __builtin_prefetch(A + (size_t)(m0 + arow) * K + (size_t)((kt + 2) * BK + acol), 0, 0);
            __builtin_prefetch(W + (size_t)((kt + 2) * BK + bk) * N + (size_t)(n0 + bnc), 0, 0);
        }

        // A frag: lane lr rows, K-half lh -> contiguous 16 f16 (32B, aligned)
        v16h af  = *(const v16h*)&As[wm * 16 + lr][lh * 16];
        v16h bf0 = *(const v16h*)&Bs[wn * 32 +      lr][lh * 16];
        v16h bf1 = *(const v16h*)&Bs[wn * 32 + 16 + lr][lh * 16];

        acc0 = __builtin_amdgcn_wmma_f32_16x16x32_f16(false, af, false, bf0,
                                                      (short)0, acc0, false, false);
        acc1 = __builtin_amdgcn_wmma_f32_16x16x32_f16(false, af, false, bf1,
                                                      (short)0, acc1, false, false);
        __syncthreads();
    }

    // Epilogue. C/D layout: VGPR r, lanes 0-15 -> M=r, N=lane; lanes 16-31 -> M=r+8.
    #pragma unroll
    for (int t = 0; t < 2; ++t) {
        const int gn = n0 + wn * 32 + t * 16 + lr;
        if (gn >= N) continue;
        #pragma unroll
        for (int r = 0; r < 8; ++r) {
            const int gm = m0 + wm * 16 + lh * 8 + r;
            const size_t idx = (size_t)gm * N + gn;
            float v = (t == 0) ? acc0[r] : acc1[r];
            if (bias)            v += bias[gn];
            if (flags & F_ACC)   v += C[idx];
            if (flags & F_BN)    v = (v - bnm[gn]) * rsqrtf(bnv[gn] + 1e-5f) * bng[gn] + bnb[gn];
            if (flags & F_RES)   v += Res[idx];
            if (flags & F_LRELU) v = v > 0.0f ? v : 0.01f * v;
            C[idx] = v;
        }
    }
}

__global__ void zero_kernel(float* __restrict__ p, size_t n)
{
    size_t i = (size_t)blockIdx.x * blockDim.x + threadIdx.x;
    if (i < n) p[i] = 0.0f;
}

// acc[dst[e], :] += feat[src[e], :] ; cnt[dst[e]] += 1   (src==null -> src=e)
__global__ void seg_scatter_add(const float* __restrict__ feat,
                                const int* __restrict__ src,
                                const int* __restrict__ dst,
                                float* __restrict__ acc, float* __restrict__ cnt,
                                int E, int D)
{
    long long i = (long long)blockIdx.x * blockDim.x + threadIdx.x;
    if (i >= (long long)E * D) return;
    const int e = (int)(i / D);
    const int d = (int)(i % D);
    const int s = src ? src[e] : e;
    const int t = dst[e];
    atomicAdd(&acc[(size_t)t * D + d], feat[(size_t)s * D + d]);
    if (d == 0) atomicAdd(&cnt[t], 1.0f);
}

__global__ void seg_div(float* __restrict__ acc, const float* __restrict__ cnt,
                        int Nrows, int D)
{
    size_t i = (size_t)blockIdx.x * blockDim.x + threadIdx.x;
    if (i >= (size_t)Nrows * D) return;
    acc[i] *= 1.0f / fmaxf(cnt[i / D], 1.0f);
}

__global__ void broadcast_rows(const float* __restrict__ gm,
                               const int* __restrict__ graph,
                               float* __restrict__ out, int NFr, int D)
{
    size_t i = (size_t)blockIdx.x * blockDim.x + threadIdx.x;
    if (i >= (size_t)NFr * D) return;
    const int f = (int)(i / D);
    const int d = (int)(i % D);
    out[i] = gm[(size_t)graph[f] * D + d];
}

// ---------------------------------------------------------------------------

extern "C" void kernel_launch(void* const* d_in, const int* in_sizes, int n_in,
                              void* d_out, int out_size, void* d_ws, size_t ws_size,
                              hipStream_t stream)
{
    (void)n_in; (void)out_size; (void)ws_size;
    const int NF = 16384, NV = 32768, Bg = 64;
    const int IN = 256, H = 512, NC = 25, FRAW = 700, VRAW = 64;
    const int EFF = in_sizes[2], EVF = in_sizes[4], EFV = in_sizes[6];

    const float* face_x   = (const float*)d_in[0];
    const float* vertex_x = (const float*)d_in[1];
    const int* ff_src = (const int*)d_in[2];
    const int* ff_dst = (const int*)d_in[3];
    const int* vf_src = (const int*)d_in[4];
    const int* vf_dst = (const int*)d_in[5];
    const int* fv_src = (const int*)d_in[6];
    const int* fv_dst = (const int*)d_in[7];
    const int* face_graph = (const int*)d_in[8];

    auto F = [&](int i) { return (const float*)d_in[i]; };
    int q = 9;
    const float* few = F(q++); const float* feb = F(q++);
    const float* v1w = F(q++); const float* v1b = F(q++);
    const float *v1g = F(q++), *v1bb = F(q++), *v1m = F(q++), *v1v = F(q++);
    const float* v2w = F(q++); const float* v2b = F(q++);
    const float *v2g = F(q++), *v2bb = F(q++), *v2m = F(q++), *v2v = F(q++);
    const float* v3w = F(q++); const float* v3b = F(q++);
    const float *c1ff_ws = F(q++), *c1ff_wn = F(q++), *c1ff_b = F(q++);
    const float *c1vf_ws = F(q++), *c1vf_wn = F(q++), *c1vf_b = F(q++);
    const float *c1fv_ws = F(q++), *c1fv_wn = F(q++), *c1fv_b = F(q++);
    const float *bn1g = F(q++), *bn1b = F(q++), *bn1m = F(q++), *bn1v = F(q++);
    const float *c2ff_ws = F(q++), *c2ff_wn = F(q++), *c2ff_b = F(q++);
    const float *c2vf_ws = F(q++), *c2vf_wn = F(q++), *c2vf_b = F(q++);
    const float *c2fv_ws = F(q++), *c2fv_wn = F(q++), *c2fv_b = F(q++);
    (void)c2fv_ws; (void)c2fv_wn; (void)c2fv_b; // h2v is dead code downstream
    const float *bn2g = F(q++), *bn2b = F(q++), *bn2m = F(q++), *bn2v = F(q++);
    const float *fw  = F(q++), *fb  = F(q++);
    const float *l1w = F(q++), *l1b = F(q++);
    const float *l1g = F(q++), *l1bb = F(q++), *l1m = F(q++), *l1v = F(q++);
    const float *l2w = F(q++), *l2b = F(q++);
    const float *l2g = F(q++), *l2bb = F(q++), *l2m = F(q++), *l2v = F(q++);
    const float *l3w = F(q++), *l3b = F(q++);

    // ---- workspace layout (floats), with region reuse (~277 MB) ----
    float* ws = (float*)d_ws;
    float* hf     = ws;                 // NF*256
    float* t1     = ws + 4194304;       // NV*128   -> reused as x1 (NF*256)
    float* t2     = ws + 8388608;       // NV*256   -> reused as neighF (NF*512)
    float* hv     = ws + 16777216;      // NV*256   -> reused as h2f (NF*512)
    float* neighV = ws + 25165824;      // NV*256   -> reused as hf_out (NF*512)
    float* h1f    = ws + 33554432;      // NF*512
    float* h1v    = ws + 41943040;      // NV*512
    float* gfeat  = ws + 58720256;      // NF*512
    float* x2     = ws + 67108864;      // NF*128
    float* gmeanb = ws + 69206016;      // B*512
    float* cntF   = ws + 69238784;      // NF
    float* cntV   = ws + 69255168;      // NV
    float* cntB   = ws + 69287936;      // B
    float* neighF = t2;
    float* h2f    = hv;
    float* hf_out = neighV;
    float* x1     = t1;

    auto gemm = [&](const float* A, const float* W, const float* bias,
                    const float* g, const float* b2, const float* m, const float* vv,
                    const float* res, float* C, int M, int N, int K, int flags) {
        dim3 grid((N + BN - 1) / BN, (M + BM - 1) / BM);
        gemm_wmma<<<grid, 256, 0, stream>>>(A, W, bias, g, b2, m, vv, res, C, M, N, K, flags);
    };
    auto segmean = [&](const float* feat, const int* src, const int* dst,
                       float* acc, float* cnt, int E, int Ndst, int D) {
        size_t an = (size_t)Ndst * D;
        zero_kernel<<<(unsigned)((an + 255) / 256), 256, 0, stream>>>(acc, an);
        zero_kernel<<<(unsigned)(((size_t)Ndst + 255) / 256), 256, 0, stream>>>(cnt, (size_t)Ndst);
        long long tot = (long long)E * D;
        seg_scatter_add<<<(unsigned)((tot + 255) / 256), 256, 0, stream>>>(feat, src, dst, acc, cnt, E, D);
        seg_div<<<(unsigned)((an + 255) / 256), 256, 0, stream>>>(acc, cnt, Ndst, D);
    };

    // ---- encoders ----
    gemm(face_x, few, feb, 0, 0, 0, 0, 0, hf, NF, IN, FRAW, 0);
    gemm(vertex_x, v1w, v1b, v1g, v1bb, v1m, v1v, 0, t1, NV, 128, VRAW, F_BN | F_LRELU);
    gemm(t1, v2w, v2b, v2g, v2bb, v2m, v2v, 0, t2, NV, 256, 128, F_BN | F_LRELU);
    gemm(t2, v3w, v3b, 0, 0, 0, 0, 0, hv, NV, IN, 256, 0);

    // ---- RGCN layer 1: faces (sum of ff + vf SAGE relations) ----
    gemm(hf, c1ff_ws, c1ff_b, 0, 0, 0, 0, 0, h1f, NF, H, IN, 0);
    segmean(hf, ff_src, ff_dst, neighF, cntF, EFF, NF, IN);
    gemm(neighF, c1ff_wn, 0, 0, 0, 0, 0, 0, h1f, NF, H, IN, F_ACC);
    gemm(hf, c1vf_ws, c1vf_b, 0, 0, 0, 0, 0, h1f, NF, H, IN, F_ACC);
    segmean(hv, vf_src, vf_dst, neighF, cntF, EVF, NF, IN);
    gemm(neighF, c1vf_wn, 0, bn1g, bn1b, bn1m, bn1v, 0, h1f, NF, H, IN,
         F_ACC | F_BN | F_LRELU);
    // layer 1: vertices (fv relation)
    gemm(hv, c1fv_ws, c1fv_b, 0, 0, 0, 0, 0, h1v, NV, H, IN, 0);
    segmean(hf, fv_src, fv_dst, neighV, cntV, EFV, NV, IN);
    gemm(neighV, c1fv_wn, 0, bn1g, bn1b, bn1m, bn1v, 0, h1v, NV, H, IN,
         F_ACC | F_BN | F_LRELU);

    // ---- RGCN layer 2: faces only (h2v never reaches the classifier) ----
    gemm(h1f, c2ff_ws, c2ff_b, 0, 0, 0, 0, 0, h2f, NF, H, H, 0);
    segmean(h1f, ff_src, ff_dst, neighF, cntF, EFF, NF, H);
    gemm(neighF, c2ff_wn, 0, 0, 0, 0, 0, 0, h2f, NF, H, H, F_ACC);
    gemm(h1f, c2vf_ws, c2vf_b, 0, 0, 0, 0, 0, h2f, NF, H, H, F_ACC);
    segmean(h1v, vf_src, vf_dst, neighF, cntF, EVF, NF, H);
    gemm(neighF, c2vf_wn, 0, bn2g, bn2b, bn2m, bn2v, h1f, h2f, NF, H, H,
         F_ACC | F_BN | F_RES | F_LRELU);

    // ---- fuse: concat([h1f,h2f]) @ Wfuse == h1f@W_top + h2f@W_bot ----
    gemm(h1f, fw, fb, 0, 0, 0, 0, 0, hf_out, NF, H, H, 0);
    gemm(h2f, fw + (size_t)H * H, 0, 0, 0, 0, 0, 0, hf_out, NF, H, H, F_ACC);

    // ---- per-graph mean over faces, broadcast back ----
    segmean(hf_out, nullptr, face_graph, gmeanb, cntB, NF, Bg, H);
    broadcast_rows<<<(unsigned)(((size_t)NF * H + 255) / 256), 256, 0, stream>>>(
        gmeanb, face_graph, gfeat, NF, H);

    // ---- classifier: concat([hf_out, gfeat]) folded into split-W GEMMs ----
    gemm(hf_out, l1w, l1b, 0, 0, 0, 0, 0, x1, NF, 256, H, 0);
    gemm(gfeat, l1w + (size_t)H * 256, 0, l1g, l1bb, l1m, l1v, 0, x1, NF, 256, H,
         F_ACC | F_BN | F_LRELU);
    gemm(x1, l2w, l2b, l2g, l2bb, l2m, l2v, 0, x2, NF, 128, 256, F_BN | F_LRELU);
    gemm(x2, l3w, l3b, 0, 0, 0, 0, 0, (float*)d_out, NF, NC, 128, 0);
}
